// PolarLinearAttention_47691316854821
// MI455X (gfx1250) — compile-verified, matrix operand order checked by reading, not run on hardware
//
#include <hip/hip_runtime.h>

typedef unsigned short u16;
typedef __attribute__((ext_vector_type(16))) __bf16 bf16x16;
typedef __attribute__((ext_vector_type(8)))  float  v8f;

union BfPack { uint4 u[2]; bf16x16 v; };

__device__ __forceinline__ unsigned f2bf(float f) {
  unsigned u = __float_as_uint(f);
  return (u + 0x7fffu + ((u >> 16) & 1u)) >> 16;   // round-to-nearest-even
}
__device__ __forceinline__ float bf2f(unsigned h) { return __uint_as_float(h << 16); }

constexpr int B_ = 8, C_ = 128, H_ = 128, W_ = 128, NH_ = 8, HD_ = 16;
constexpr int N_ = H_ * W_;          // 16384 pixels per image
constexpr int M_ = B_ * N_;          // 131072 rows

// ---------------------------------------------------------------- kernel 0
__global__ void k_wconv(const float* __restrict__ qkv_w, const float* __restrict__ proj_w,
                        u16* __restrict__ qkvw, u16* __restrict__ projw) {
  int i = blockIdx.x * blockDim.x + threadIdx.x;
  if (i < 3 * C_ * C_) qkvw[i] = (u16)f2bf(qkv_w[i]);
  if (i < C_ * C_)     projw[i] = (u16)f2bf(proj_w[i]);
}

// ------------------------------------------------- kernel 1: polar+conv+LN
// one wave per pixel; lane handles 4 consecutive channels
__global__ void k_pre(const float* __restrict__ x, const float* __restrict__ conv_w,
                      const float* __restrict__ conv_b, const float* __restrict__ ln_g,
                      const float* __restrict__ ln_b, u16* __restrict__ xf) {
  const int lane = threadIdx.x & 31;
  const int pid  = blockIdx.x * (blockDim.x >> 5) + (threadIdx.x >> 5);
  const int b = pid / N_;
  const int n = pid % N_;
  const int h = n / W_, w = n % W_;
  const int c0 = lane * 4;

  float wgt[4][9];
#pragma unroll
  for (int j = 0; j < 4; ++j)
#pragma unroll
    for (int t = 0; t < 9; ++t) wgt[j][t] = conv_w[(c0 + j) * 9 + t];

  float acc[4] = {conv_b[c0], conv_b[c0 + 1], conv_b[c0 + 2], conv_b[c0 + 3]};
  const float PI = 3.14159265358979323846f;
#pragma unroll
  for (int t = 0; t < 9; ++t) {
    const int ky = t / 3 - 1, kx = t % 3 - 1;
    const int hh = h + ky, ww = w + kx;
    if (hh < 0 || hh >= H_ || ww < 0 || ww >= W_) continue;   // SAME zero-pad
    const float fy = (float)(hh - 64), fx = (float)(ww - 64);
    const float rho = sqrtf(fx * fx + fy * fy) * (1.0f / 64.0f);
    const float th  = (atan2f(fy, fx) + PI) * (0.5f / PI);
    const long base = ((long)b * C_ + c0) * N_ + hh * W_ + ww;
#pragma unroll
    for (int j = 0; j < 4; ++j) {
      const float pv = ((c0 + j) & 1) ? th : rho;
      acc[j] += (x[base + (long)j * N_] + pv) * wgt[j][t];
    }
  }
  // LayerNorm over C=128 (wave allreduce)
  float s  = acc[0] + acc[1] + acc[2] + acc[3];
  float sq = acc[0]*acc[0] + acc[1]*acc[1] + acc[2]*acc[2] + acc[3]*acc[3];
#pragma unroll
  for (int off = 16; off; off >>= 1) {
    s  += __shfl_xor(s,  off, 32);
    sq += __shfl_xor(sq, off, 32);
  }
  const float mu   = s * (1.0f / 128.0f);
  const float var  = sq * (1.0f / 128.0f) - mu * mu;
  const float rinv = rsqrtf(var + 1e-5f);
  unsigned p[4];
#pragma unroll
  for (int j = 0; j < 4; ++j)
    p[j] = f2bf((acc[j] - mu) * rinv * ln_g[c0 + j] + ln_b[c0 + j]);
  uint2 o; o.x = p[0] | (p[1] << 16); o.y = p[2] | (p[3] << 16);
  *(uint2*)(xf + (size_t)pid * C_ + c0) = o;
}

// ----------------------------------------------------- kernel 2: QKV GEMM
// one wave = 16 pixels x 128 outputs of one t in {q,k,v}: 8 j-tiles (= 8 heads),
// A-chunk loaded once per K-step and reused across all 8 WMMAs.
__global__ void k_qkv(const u16* __restrict__ xf, const u16* __restrict__ qkvw,
                      u16* __restrict__ qA, u16* __restrict__ kT, u16* __restrict__ vT) {
  const int lane = threadIdx.x & 31;
  const int wid  = blockIdx.x * (blockDim.x >> 5) + (threadIdx.x >> 5);
  const int tsel = wid % 3;                               // 0=q, 1=k, 2=v
  const int mt   = wid / 3;
  const int m0 = mt * 16;
  const int b  = m0 / N_, n0 = m0 % N_;
  const int half = lane >> 4, l16 = lane & 15, koff = half * 8;
  const int row = m0 + l16;

  v8f acc[8];
#pragma unroll
  for (int jt = 0; jt < 8; ++jt) acc[jt] = (v8f){0.f,0.f,0.f,0.f,0.f,0.f,0.f,0.f};

#pragma unroll
  for (int kc = 0; kc < 4; ++kc) {
    BfPack a;
    const uint4* pa = (const uint4*)(xf + (size_t)row * C_ + kc * 32 + koff);
    a.u[0] = pa[0]; a.u[1] = pa[2];                       // K 0-7 / 16-23 (lane half)
#pragma unroll
    for (int jt = 0; jt < 8; ++jt) {
      BfPack bm;
      const uint4* pb =
          (const uint4*)(qkvw + (size_t)(tsel * C_ + jt * 16 + l16) * C_ + kc * 32 + koff);
      bm.u[0] = pb[0]; bm.u[1] = pb[2];
      acc[jt] = __builtin_amdgcn_wmma_f32_16x16x32_bf16(false, a.v, false, bm.v,
                                                        (short)0, acc[jt], false, false);
    }
  }

  if (tsel == 0) {
    // q: relu, scatter to pixel-major [m,128]
#pragma unroll
    for (int jt = 0; jt < 8; ++jt)
#pragma unroll
      for (int r = 0; r < 8; ++r) {
        const int m = r + half * 8;
        qA[((size_t)(m0 + m)) * C_ + jt * 16 + l16] = (u16)f2bf(fmaxf(acc[jt][r], 0.0f));
      }
  } else {
    u16* dst = (tsel == 1) ? kT : vT;
#pragma unroll
    for (int jt = 0; jt < 8; ++jt) {                      // jt == head, l16 == d
      unsigned q_[8];
#pragma unroll
      for (int r = 0; r < 8; ++r) {
        float v = acc[jt][r];
        if (tsel == 1) v = fmaxf(v, 0.0f);                // relu on k
        q_[r] = f2bf(v);
      }
      uint4 pk;
      pk.x = q_[0] | (q_[1] << 16); pk.y = q_[2] | (q_[3] << 16);
      pk.z = q_[4] | (q_[5] << 16); pk.w = q_[6] | (q_[7] << 16);
      const size_t off = ((size_t)((b * NH_ + jt) * HD_ + l16)) * N_ + n0 + half * 8;
      *(uint4*)(dst + off) = pk;                          // 8 contiguous pixels
    }
  }
}

// ---------------------------------------- kernel 3: context = k^T v + k_sum
// one block (4 waves) per (b,head); 512 n-chunks of 32; prefetch next chunk
__global__ void k_ctx(const u16* __restrict__ kT, const u16* __restrict__ vT,
                      u16* __restrict__ ctxT, float* __restrict__ ksum) {
  __shared__ float cs[4][8][32];
  __shared__ float ks[4][32];
  const int lane = threadIdx.x & 31;
  const int wv   = threadIdx.x >> 5;
  const int bh   = blockIdx.x;                            // 0..63
  const int half = lane >> 4, l16 = lane & 15, koff = half * 8;
  const u16* kp = kT + ((size_t)(bh * HD_ + l16)) * N_;   // A lane row = d
  const u16* vp = vT + ((size_t)(bh * HD_ + l16)) * N_;   // B lane col = e

  v8f acc = {0.f,0.f,0.f,0.f,0.f,0.f,0.f,0.f};
  float kacc = 0.f;
  for (int ch = wv; ch < N_ / 32; ch += 4) {
    const int n0 = ch * 32;
    __builtin_prefetch((const void*)(kp + n0 + 128 + koff), 0, 1);  // next iter
    __builtin_prefetch((const void*)(vp + n0 + 128 + koff), 0, 1);
    BfPack a, bm;
    const uint4* pk = (const uint4*)(kp + n0 + koff);
    a.u[0] = pk[0]; a.u[1] = pk[2];
    const uint4* pv = (const uint4*)(vp + n0 + koff);
    bm.u[0] = pv[0]; bm.u[1] = pv[2];
    const unsigned aw0[4] = {a.u[0].x, a.u[0].y, a.u[0].z, a.u[0].w};
    const unsigned aw1[4] = {a.u[1].x, a.u[1].y, a.u[1].z, a.u[1].w};
#pragma unroll
    for (int i = 0; i < 4; ++i)
      kacc += bf2f(aw0[i] & 0xffffu) + bf2f(aw0[i] >> 16) +
              bf2f(aw1[i] & 0xffffu) + bf2f(aw1[i] >> 16);
    acc = __builtin_amdgcn_wmma_f32_16x16x32_bf16(false, a.v, false, bm.v,
                                                  (short)0, acc, false, false);
  }
  kacc += __shfl_xor(kacc, 16, 32);                       // combine lane halves (same d)
#pragma unroll
  for (int r = 0; r < 8; ++r) cs[wv][r][lane] = acc[r];
  ks[wv][lane] = kacc;
  __syncthreads();
  if (wv == 0) {
    float tot[8] = {0,0,0,0,0,0,0,0};
    float kt = 0.f;
#pragma unroll
    for (int w = 0; w < 4; ++w) {
#pragma unroll
      for (int r = 0; r < 8; ++r) tot[r] += cs[w][r][lane];
      kt += ks[w][lane];
    }
    unsigned p[8];
#pragma unroll
    for (int r = 0; r < 8; ++r) p[r] = f2bf(tot[r]);
    uint4 o;
    o.x = p[0] | (p[1] << 16); o.y = p[2] | (p[3] << 16);
    o.z = p[4] | (p[5] << 16); o.w = p[6] | (p[7] << 16);
    // ctxT[e][d] (e-major): lane = e, d = half*8 + r  -> contiguous 16B
    *(uint4*)(ctxT + ((size_t)(bh * HD_ + l16)) * HD_ + half * 8) = o;
    if (lane < 16) ksum[bh * HD_ + lane] = kt;
  }
}

// -------------------------- kernel 4: attention out + head concat + proj + res
// one wave = one 16-pixel tile; LDS transposes head outputs to row-major,
// then the 4 A-chunks are hoisted to registers for the 8-column proj loop.
__global__ void k_out(const u16* __restrict__ qA, const u16* __restrict__ ctxT,
                      const float* __restrict__ ksum, const u16* __restrict__ projw,
                      const float* __restrict__ proj_b, const float* __restrict__ x,
                      float* __restrict__ out) {
  __shared__ u16 attn[4][16 * 128];
  const int lane = threadIdx.x & 31;
  const int wv   = threadIdx.x >> 5;
  const int tid  = blockIdx.x * (blockDim.x >> 5) + wv;   // 0..8191
  const int m0 = tid * 16;
  const int b  = m0 / N_, n0 = m0 % N_;
  const int half = lane >> 4, l16 = lane & 15, koff = half * 8;
  u16* at = attn[wv];
  const uint4 zero4 = {0u, 0u, 0u, 0u};

  for (int hh = 0; hh < NH_; ++hh) {
    const int bh = b * NH_ + hh;
    BfPack a, bm;
    const uint4* pq = (const uint4*)(qA + ((size_t)(m0 + l16)) * C_ + hh * HD_ + koff);
    a.u[0] = pq[0]; a.u[1] = zero4;                       // K (=d) padded 16->32
    const uint4* pc = (const uint4*)(ctxT + ((size_t)(bh * HD_ + l16)) * HD_ + koff);
    bm.u[0] = pc[0]; bm.u[1] = zero4;
    // d_inv = 1/max(q . k_sum, 1e-6)
    const float4 kv0 = *(const float4*)(ksum + bh * HD_ + koff);
    const float4 kv1 = *(const float4*)(ksum + bh * HD_ + koff + 4);
    const float kk[8] = {kv0.x, kv0.y, kv0.z, kv0.w, kv1.x, kv1.y, kv1.z, kv1.w};
    const unsigned aw[4] = {a.u[0].x, a.u[0].y, a.u[0].z, a.u[0].w};
    float dot = 0.f;
#pragma unroll
    for (int i = 0; i < 4; ++i)
      dot += bf2f(aw[i] & 0xffffu) * kk[2 * i] + bf2f(aw[i] >> 16) * kk[2 * i + 1];
    dot += __shfl_xor(dot, 16, 32);                       // full dot for row l16
    const float dinv = 1.0f / fmaxf(dot, 1e-6f);
    v8f o = {0.f,0.f,0.f,0.f,0.f,0.f,0.f,0.f};
    o = __builtin_amdgcn_wmma_f32_16x16x32_bf16(false, a.v, false, bm.v,
                                                (short)0, o, false, false);
#pragma unroll
    for (int r = 0; r < 8; ++r) {
      const float dv = __shfl(dinv, r + half * 8, 32);    // d_inv of row m
      const int m = r + half * 8;
      at[m * C_ + hh * HD_ + l16] = (u16)f2bf(o[r] * dv);
    }
  }
  __syncthreads();

  // hoist the 4 LDS A-chunks (16x128 row-major attn tile) into registers
  BfPack aa[4];
#pragma unroll
  for (int kc = 0; kc < 4; ++kc) {
    const uint4* pa = (const uint4*)(at + l16 * C_ + kc * 32 + koff);
    aa[kc].u[0] = pa[0]; aa[kc].u[1] = pa[2];
  }

  // projection: [16x128] @ proj_w^T, + bias + residual, NCHW store
#pragma unroll
  for (int ct = 0; ct < 8; ++ct) {
    v8f acc = {0.f,0.f,0.f,0.f,0.f,0.f,0.f,0.f};
#pragma unroll
    for (int kc = 0; kc < 4; ++kc) {
      BfPack bm;
      const uint4* pb = (const uint4*)(projw + (size_t)(ct * 16 + l16) * C_ + kc * 32 + koff);
      bm.u[0] = pb[0]; bm.u[1] = pb[2];
      acc = __builtin_amdgcn_wmma_f32_16x16x32_bf16(false, aa[kc].v, false, bm.v,
                                                    (short)0, acc, false, false);
    }
    const int co = ct * 16 + l16;
    const float pb = proj_b[co];
    const size_t base = ((size_t)(b * C_ + co)) * N_ + n0 + half * 8;
    const float4 x0 = *(const float4*)(x + base);
    const float4 x1 = *(const float4*)(x + base + 4);
    float4 o0, o1;
    o0.x = acc[0] + pb + x0.x; o0.y = acc[1] + pb + x0.y;
    o0.z = acc[2] + pb + x0.z; o0.w = acc[3] + pb + x0.w;
    o1.x = acc[4] + pb + x1.x; o1.y = acc[5] + pb + x1.y;
    o1.z = acc[6] + pb + x1.z; o1.w = acc[7] + pb + x1.w;
    *(float4*)(out + base)     = o0;
    *(float4*)(out + base + 4) = o1;
  }
}

// ---------------------------------------------------------------- launcher
extern "C" void kernel_launch(void* const* d_in, const int* in_sizes, int n_in,
                              void* d_out, int out_size, void* d_ws, size_t ws_size,
                              hipStream_t stream) {
  const float* x      = (const float*)d_in[0];
  const float* qkv_w  = (const float*)d_in[1];
  const float* proj_w = (const float*)d_in[2];
  const float* proj_b = (const float*)d_in[3];
  const float* conv_w = (const float*)d_in[4];
  const float* conv_b = (const float*)d_in[5];
  const float* ln_g   = (const float*)d_in[6];
  const float* ln_b   = (const float*)d_in[7];
  float* out = (float*)d_out;

  char* ws = (char*)d_ws;
  size_t off = 0;
  auto alloc = [&](size_t bytes) {
    char* p = ws + off;
    off += (bytes + 255) & ~(size_t)255;
    return p;
  };
  u16*   xf   = (u16*)alloc((size_t)M_ * C_ * 2);        // LN'd features, bf16
  u16*   qA   = (u16*)alloc((size_t)M_ * C_ * 2);        // q, pixel-major
  u16*   kT   = (u16*)alloc((size_t)M_ * C_ * 2);        // k^T [b,h,d,N]
  u16*   vT   = (u16*)alloc((size_t)M_ * C_ * 2);        // v^T [b,h,e,N]
  u16*   ctxT = (u16*)alloc((size_t)B_ * NH_ * HD_ * HD_ * 2);
  float* ksm  = (float*)alloc((size_t)B_ * NH_ * HD_ * 4);
  u16*   qkvw = (u16*)alloc((size_t)3 * C_ * C_ * 2);
  u16*   prjw = (u16*)alloc((size_t)C_ * C_ * 2);
  (void)in_sizes; (void)n_in; (void)out_size; (void)ws_size;

  k_wconv<<<(3 * C_ * C_ + 255) / 256, 256, 0, stream>>>(qkv_w, proj_w, qkvw, prjw);
  k_pre  <<<M_ / 8, 256, 0, stream>>>(x, conv_w, conv_b, ln_g, ln_b, xf);
  k_qkv  <<<(M_ / 16) * 3 / 8, 256, 0, stream>>>(xf, qkvw, qA, kT, vT);
  k_ctx  <<<B_ * NH_, 128, 0, stream>>>(kT, vT, ctxT, ksm);
  k_out  <<<(M_ / 16) / 4, 128, 0, stream>>>(qA, ctxT, ksm, prjw, proj_b, x, out);
}
